// Seq2SeqDifference_85667417686023
// MI455X (gfx1250) — matrix operand order repaired and checked by reading, not compiled, not run on hardware
//
#include <hip/hip_runtime.h>
#include <hip/hip_bf16.h>

typedef unsigned short u16;
typedef __bf16 v16bf __attribute__((ext_vector_type(16)));
typedef float  v8f   __attribute__((ext_vector_type(8)));
typedef unsigned int v4u __attribute__((ext_vector_type(4)));
typedef int v8i __attribute__((ext_vector_type(8)));
typedef int v4i __attribute__((ext_vector_type(4)));

static constexpr int Bn = 64, Ln = 512, Rn = 512, Dn = 512;
static constexpr int Fc = 100, Fp = 112, Hn = 60, TwoF = 200;
static constexpr int BT_STRIDE = 520;   // 512 bf16 + 8 (16B TDM pad) per row

#if defined(__has_builtin)
#  if __has_builtin(__builtin_amdgcn_tensor_load_to_lds)
#    define HAVE_TDM 1
#  endif
#endif
#ifndef HAVE_TDM
#  define HAVE_TDM 0
#endif

union ABu { v16bf v; uint4 q[2]; };

// ---------- helpers ----------
__device__ __forceinline__ u16 f2bf(float x) {
  unsigned int u = __float_as_uint(x);
  unsigned int r = u + 0x7FFFu + ((u >> 16) & 1u);   // round-to-nearest-even
  return (u16)(r >> 16);
}
__device__ __forceinline__ float wred_max(float x) {
  for (int m = 16; m; m >>= 1) x = fmaxf(x, __shfl_xor(x, m, 32));
  return x;
}
__device__ __forceinline__ float wred_sum(float x) {
  for (int m = 16; m; m >>= 1) x += __shfl_xor(x, m, 32);
  return x;
}
__device__ __forceinline__ void wred_argmax(float& v, int& i) {
  for (int m = 16; m; m >>= 1) {
    float ov = __shfl_xor(v, m, 32);
    int   oi = __shfl_xor(i, m, 32);
    if (ov > v || (ov == v && oi < i)) { v = ov; i = oi; }
  }
}

// ---------- 1) conversions / init ----------
__global__ void k_convert_vec(const float* __restrict__ src, u16* __restrict__ dst, int S) {
  int idx = blockIdx.x * 256 + threadIdx.x;              // total = S*B*D
  int d = idx & (Dn - 1);
  int rest = idx >> 9;
  int b = rest & (Bn - 1);
  int s = rest >> 6;
  if (s >= S) return;
  dst[((size_t)b * S + s) * Dn + d] = f2bf(src[((size_t)s * Bn + b) * Dn + d]);
}
__global__ void k_convert_wconv(const float* __restrict__ w, u16* __restrict__ wb) {
  int idx = blockIdx.x * 256 + threadIdx.x;              // total = 112*512
  if (idx >= Fp * Dn) return;
  int f = idx >> 9, d = idx & (Dn - 1);
  wb[idx] = f2bf(f < Fc ? w[f * Dn + d] : 0.0f);
}
__global__ void k_zero(float* __restrict__ p, int n) {
  int i = blockIdx.x * 256 + threadIdx.x;
  if (i < n) p[i] = 0.0f;
}

// ---------- 2) batched masked attention GEMM (bf16 WMMA, TDM-staged B) ----------
// Block: 128 threads = 4 waves, computes a 128(L) x 32(R) output tile for batch b.
// B tile (32 rows x 512 d, bf16) is staged Global->LDS by the Tensor Data Mover
// with 16B row padding (stride 1040B) for bank-friendly B-fragment reads.
__global__ void k_attention(const u16* __restrict__ lbf, const u16* __restrict__ rbf,
                            const int* __restrict__ llen, const int* __restrict__ rlen,
                            float* __restrict__ att) {
  __shared__ __align__(16) u16 bTile[32 * BT_STRIDE];    // 33,280 B
  const int wave = threadIdx.x >> 5, lane = threadIdx.x & 31;
  const int b = blockIdx.z;
  const int r0 = blockIdx.x * 32;
  const int l0 = blockIdx.y * 128 + wave * 32;
  const int m = lane & 15, n = lane & 15, kh = lane >> 4;

  const unsigned long long gB =
      (unsigned long long)(const void*)(rbf + ((size_t)b * Rn + r0) * Dn);

#if HAVE_TDM
  if (wave == 0) {
    unsigned int ldsOff = (unsigned int)(size_t)(void*)&bTile[0];
    v4u g0;
    g0.x = 1u;                                           // count=1, user descriptor
    g0.y = ldsOff;                                       // lds_addr
    g0.z = (unsigned int)(gB & 0xFFFFFFFFull);           // global_addr[31:0]
    g0.w = (unsigned int)((gB >> 32) & 0x01FFFFFFull) | (2u << 30);  // addr hi | type=2
    v8i g1;
    g1[0] = (1 << 16) | (1 << 20) | (7 << 22) | (3 << 25); // data_size=2B, pad_en, 1KB interval, 16B pad
    g1[1] = (Dn & 0xFFFF) << 16;                         // tensor_dim0 = 512
    g1[2] = (32 << 16);                                  // tensor_dim1 = 32 rows
    g1[3] = (Dn & 0xFFFF) << 16;                         // tile_dim0 = 512
    g1[4] = 32;                                          // tile_dim1 = 32, tile_dim2 = 0
    g1[5] = Dn;                                          // tensor_dim0_stride = 512
    g1[6] = 0;
    g1[7] = 0;
    v4i gz = {0, 0, 0, 0};
#  if __has_include(<hip/amd_detail/amd_gfx1250_TDM.h>)
    v8i gz8 = {0, 0, 0, 0, 0, 0, 0, 0};
    __builtin_amdgcn_tensor_load_to_lds(g0, g1, gz, gz, gz8, 0);   // clang-23 / therock form
#  else
    __builtin_amdgcn_tensor_load_to_lds(g0, g1, gz, gz, 0);        // ROCm 7.2 form
#  endif
    __builtin_amdgcn_s_wait_tensorcnt(0);
  }
#else
  // fallback: cooperative copy with padding
  {
    const uint4* src = (const uint4*)(const void*)gB;    // 32*512 u16 = 2048 uint4
    for (int i = threadIdx.x; i < 2048; i += 128) {
      int row = i >> 6, col = i & 63;                    // 64 uint4 per row
      *(uint4*)&bTile[row * BT_STRIDE + col * 8] = src[i];
    }
  }
#endif
  __syncthreads();

  v8f acc00 = {}, acc01 = {}, acc10 = {}, acc11 = {};
  const u16* la0 = lbf + ((size_t)b * Ln + l0 + m) * Dn;
  const u16* la1 = la0 + (size_t)16 * Dn;
  const u16* lb0 = &bTile[n * BT_STRIDE];
  const u16* lb1 = &bTile[(n + 16) * BT_STRIDE];

#pragma unroll 4
  for (int k0 = 0; k0 < Dn; k0 += 32) {
    ABu a0, a1, b0, b1;
    a0.q[0] = *(const uint4*)(la0 + k0 + kh * 8);
    a0.q[1] = *(const uint4*)(la0 + k0 + 16 + kh * 8);
    a1.q[0] = *(const uint4*)(la1 + k0 + kh * 8);
    a1.q[1] = *(const uint4*)(la1 + k0 + 16 + kh * 8);
    b0.q[0] = *(const uint4*)(lb0 + k0 + kh * 8);
    b0.q[1] = *(const uint4*)(lb0 + k0 + 16 + kh * 8);
    b1.q[0] = *(const uint4*)(lb1 + k0 + kh * 8);
    b1.q[1] = *(const uint4*)(lb1 + k0 + 16 + kh * 8);
    acc00 = __builtin_amdgcn_wmma_f32_16x16x32_bf16(false, a0.v, false, b0.v, (short)0, acc00, false, false);
    acc01 = __builtin_amdgcn_wmma_f32_16x16x32_bf16(false, a0.v, false, b1.v, (short)0, acc01, false, false);
    acc10 = __builtin_amdgcn_wmma_f32_16x16x32_bf16(false, a1.v, false, b0.v, (short)0, acc10, false, false);
    acc11 = __builtin_amdgcn_wmma_f32_16x16x32_bf16(false, a1.v, false, b1.v, (short)0, acc11, false, false);
  }

  const int ll_b = llen[b], rl_b = rlen[b];
  const float rm0 = ((r0 + n) < rl_b) ? 1.0f : 0.0f;
  const float rm1 = ((r0 + 16 + n) < rl_b) ? 1.0f : 0.0f;
#pragma unroll
  for (int v = 0; v < 8; ++v) {
    int ll0v = l0 + v + 8 * kh;
    int ll1v = ll0v + 16;
    float lm0 = (ll0v < ll_b) ? 1.0f : 0.0f;
    float lm1 = (ll1v < ll_b) ? 1.0f : 0.0f;
    float* row0 = att + ((size_t)b * Ln + ll0v) * Rn + r0 + n;
    float* row1 = att + ((size_t)b * Ln + ll1v) * Rn + r0 + n;
    row0[0]  = acc00[v] * lm0 * rm0;
    row0[16] = acc01[v] * lm0 * rm1;
    row1[0]  = acc10[v] * lm1 * rm0;
    row1[16] = acc11[v] * lm1 * rm1;
  }
}

// ---------- 3) row softmax (axis=2 over R) -> l_w, l_alpha, l_idx ----------
__global__ void k_row_softmax(const float* __restrict__ att,
                              const int* __restrict__ llen, const int* __restrict__ rlen,
                              float* __restrict__ lW, float* __restrict__ lA,
                              int* __restrict__ lI) {
  const int wid = threadIdx.x >> 5, lane = threadIdx.x & 31;
  const int row = blockIdx.x * 8 + wid;      // row = b*L + l
  const int b = row >> 9, l = row & (Ln - 1);
  const float* p = att + (size_t)row * Rn;
  const int rl = rlen[b], llv = llen[b];

  float lg[16];
  float mx = -3.4e38f;
#pragma unroll
  for (int j = 0; j < 16; ++j) {
    int r = lane + 32 * j;
    float g = p[r] - ((r < rl) ? 0.0f : 10.0f);
    lg[j] = g;
    mx = fmaxf(mx, g);
  }
  mx = wred_max(mx);

  float se = 0.f, sm = 0.f, sq = 0.f, emax = -1.f;
  int eidx = 0x7FFFFFFF;
#pragma unroll
  for (int j = 0; j < 16; ++j) {
    int r = lane + 32 * j;
    float e = __expf(lg[j] - mx);
    se += e;
    if (r < rl) {
      sm += e; sq += e * e;
      if (e > emax) { emax = e; eidx = r; }
    }
  }
  se = wred_sum(se); sm = wred_sum(sm); sq = wred_sum(sq);
  wred_argmax(emax, eidx);

  if (lane == 0) {
    float inv = 1.0f / (sm + 1e-13f * se);
    float rlf = (float)rl;
    float mean = (sm * inv) / rlf;
    float w = (sq * inv * inv) / rlf - mean * mean;
    w /= fmaxf(mean, 0.001f);
    float a = emax * inv;
    lW[row] = w;
    lA[row] = a / fmaxf(a, 0.001f);
    lI[row] = (l < llv) ? eidx : 0;
  }
}

// ---------- 4) column softmax (axis=1 over L) -> r_w, r_alpha, r_idx ----------
__global__ void k_col_softmax(const float* __restrict__ att,
                              const int* __restrict__ llen, const int* __restrict__ rlen,
                              float* __restrict__ rW, float* __restrict__ rA,
                              int* __restrict__ rI) {
  const int b = blockIdx.y;
  const int r = blockIdx.x * 256 + threadIdx.x;
  const float* base = att + (size_t)b * Ln * Rn + r;
  const int ll = llen[b], rl = rlen[b];

  float mx = -3.4e38f;
  for (int l = 0; l < Ln; ++l) {
    float g = base[(size_t)l * Rn] - ((l < ll) ? 0.0f : 10.0f);
    mx = fmaxf(mx, g);
  }
  float se = 0.f, sm = 0.f, sq = 0.f, emax = -1.f;
  int eidx = 0;
  for (int l = 0; l < Ln; ++l) {
    float g = base[(size_t)l * Rn] - ((l < ll) ? 0.0f : 10.0f);
    float e = __expf(g - mx);
    se += e;
    if (l < ll) {
      sm += e; sq += e * e;
      if (e > emax) { emax = e; eidx = l; }
    }
  }
  float inv = 1.0f / (sm + 1e-13f * se);
  float llf = (float)ll;
  float mean = (sm * inv) / llf;
  float w = (sq * inv * inv) / llf - mean * mean;
  w /= fmaxf(mean, 0.001f);
  float a = emax * inv;
  rW[b * Rn + r] = w;
  rA[b * Rn + r] = a / fmaxf(a, 0.001f);
  rI[b * Rn + r] = (r < rl) ? eidx : 0;
}

// ---------- 5/6) fused feature-build + conv GEMM + masked max-pool ----------
__global__ void k_feat_conv_pool(const float* __restrict__ xvec,   // [S,B,D] own side
                                 const float* __restrict__ gvec,   // [S2,B,D] other side
                                 const float* __restrict__ wArr,   // [B*S]
                                 const float* __restrict__ aArr,   // [B*S]
                                 const int*   __restrict__ iArr,   // [B*S]
                                 const int*   __restrict__ len,    // [B]
                                 const u16*   __restrict__ wconv,  // [112,512] bf16
                                 const float* __restrict__ bconv,  // [100]
                                 float* __restrict__ pooled,       // [B,200]
                                 int S, int poolOff) {
  __shared__ __align__(32) u16 cTile[16 * Dn];
  const int wave = threadIdx.x >> 5, lane = threadIdx.x & 31;
  const int b = blockIdx.y;
  const int s0 = blockIdx.x * 16;
  const int myLen = len[b];

  for (int i = 0; i < 4; ++i) {
    int row = wave * 4 + i;
    int s = s0 + row;
    bool valid = s < myLen;
    float wv = wArr[b * S + s];
    float al = aArr[b * S + s];
    int gi = iArr[b * S + s];
    const float* xp = xvec + ((size_t)s * Bn + b) * Dn;
    const float* gp = gvec + ((size_t)gi * Bn + b) * Dn;
    for (int d = lane; d < Dn; d += 32) {
      float c = valid ? wv * fabsf(xp[d] - al * gp[d]) : 0.0f;
      cTile[row * Dn + d] = f2bf(c);
    }
  }
  __syncthreads();

  const int m = lane & 15, n = lane & 15, kh = lane >> 4;
  for (int nt = wave; nt < 7; nt += 4) {
    const int f = nt * 16 + n;
    const u16* wb = wconv + (size_t)f * Dn;
    v8f acc = {};
#pragma unroll 4
    for (int k0 = 0; k0 < Dn; k0 += 32) {
      ABu a, bm;
      a.q[0]  = *(const uint4*)&cTile[m * Dn + k0 + kh * 8];
      a.q[1]  = *(const uint4*)&cTile[m * Dn + k0 + 16 + kh * 8];
      bm.q[0] = *(const uint4*)(wb + k0 + kh * 8);
      bm.q[1] = *(const uint4*)(wb + k0 + 16 + kh * 8);
      acc = __builtin_amdgcn_wmma_f32_16x16x32_bf16(false, a.v, false, bm.v,
                                                    (short)0, acc, false, false);
    }
    float bc = (f < Fc) ? bconv[f] : 0.0f;
#pragma unroll
    for (int v = 0; v < 8; ++v) {
      int s = s0 + v + 8 * kh;
      if (s < myLen && f < Fc) {
        float h = acc[v] + bc;
        h = h > 0.0f ? h : 0.0f;           // relu; >= 0 so int atomicMax preserves order
        atomicMax((int*)&pooled[(size_t)b * TwoF + poolOff + f], __float_as_int(h));
      }
    }
  }
}

// ---------- 7) final dense 200 -> 60 + relu ----------
__global__ void k_dense(const float* __restrict__ pooled, const float* __restrict__ Wd,
                        const float* __restrict__ bd, float* __restrict__ out) {
  int tid = blockIdx.x * 256 + threadIdx.x;
  if (tid >= Bn * Hn) return;
  int b = tid / Hn, h = tid % Hn;
  float s = bd[h];
  const float* p = pooled + (size_t)b * TwoF;
  const float* w = Wd + (size_t)h * TwoF;
#pragma unroll 8
  for (int k = 0; k < TwoF; ++k) s += p[k] * w[k];
  out[b * Hn + h] = s > 0.0f ? s : 0.0f;
}

// ---------- host ----------
extern "C" void kernel_launch(void* const* d_in, const int* in_sizes, int n_in,
                              void* d_out, int out_size, void* d_ws, size_t ws_size,
                              hipStream_t stream) {
  const int*   l_len = (const int*)d_in[0];
  const float* l_vec = (const float*)d_in[1];
  const int*   r_len = (const int*)d_in[2];
  const float* r_vec = (const float*)d_in[3];
  const float* Wconv = (const float*)d_in[4];
  const float* bconv = (const float*)d_in[5];
  const float* Wd    = (const float*)d_in[6];
  const float* bd    = (const float*)d_in[7];
  float* out = (float*)d_out;

  char* ws = (char*)d_ws;
  size_t off = 0;
  auto take = [&](size_t bytes) { size_t o = off; off = (off + bytes + 255) & ~(size_t)255; return o; };
  u16*   lbf  = (u16*)  (ws + take((size_t)Bn * Ln * Dn * 2));
  u16*   rbf  = (u16*)  (ws + take((size_t)Bn * Rn * Dn * 2));
  float* att  = (float*)(ws + take((size_t)Bn * Ln * Rn * 4));
  u16*   wcv  = (u16*)  (ws + take((size_t)Fp * Dn * 2));
  float* lW   = (float*)(ws + take((size_t)Bn * Ln * 4));
  float* lA   = (float*)(ws + take((size_t)Bn * Ln * 4));
  int*   lI   = (int*)  (ws + take((size_t)Bn * Ln * 4));
  float* rW   = (float*)(ws + take((size_t)Bn * Rn * 4));
  float* rA   = (float*)(ws + take((size_t)Bn * Rn * 4));
  int*   rI   = (int*)  (ws + take((size_t)Bn * Rn * 4));
  float* pool = (float*)(ws + take((size_t)Bn * TwoF * 4));

  {
    int total = Ln * Bn * Dn;
    k_convert_vec<<<(total + 255) / 256, 256, 0, stream>>>(l_vec, lbf, Ln);
    k_convert_vec<<<(total + 255) / 256, 256, 0, stream>>>(r_vec, rbf, Rn);
    k_convert_wconv<<<(Fp * Dn + 255) / 256, 256, 0, stream>>>(Wconv, wcv);
    k_zero<<<(Bn * TwoF + 255) / 256, 256, 0, stream>>>(pool, Bn * TwoF);
  }
  k_attention<<<dim3(Rn / 32, Ln / 128, Bn), 128, 0, stream>>>(lbf, rbf, l_len, r_len, att);
  k_row_softmax<<<(Bn * Ln) / 8, 256, 0, stream>>>(att, l_len, r_len, lW, lA, lI);
  k_col_softmax<<<dim3(Rn / 256, Bn), 256, 0, stream>>>(att, l_len, r_len, rW, rA, rI);
  k_feat_conv_pool<<<dim3(Ln / 16, Bn), 128, 0, stream>>>(l_vec, r_vec, lW, lA, lI, l_len,
                                                          wcv, bconv, pool, Ln, 0);
  k_feat_conv_pool<<<dim3(Rn / 16, Bn), 128, 0, stream>>>(r_vec, l_vec, rW, rA, rI, r_len,
                                                          wcv, bconv, pool, Rn, Fc);
  k_dense<<<(Bn * Hn + 255) / 256, 256, 0, stream>>>(pool, Wd, bd, out);
}
